// IMU_NN_52097953300620
// MI455X (gfx1250) — matrix operand order repaired and checked by reading, not compiled, not run on hardware
//
#include <hip/hip_runtime.h>

// Model constants (from reference setup_inputs)
#define L_T   32
#define B_T   2
#define D_T   64
#define HW_T  256
#define HD_T  4
#define DK_T  16
#define FF_T  128

typedef _Float16 v16h __attribute__((ext_vector_type(16)));
typedef float    v8f  __attribute__((ext_vector_type(8)));

union H16 { v16h v; uint4 q[2]; };

__device__ __forceinline__ float leakyf(float v) { return v >= 0.f ? v : 0.01f * v; }

// ---------------------------------------------------------------------------
// Feature generator stage: conv3x3(pad1)+bias -> BN (batch stats over B,H,W of
// THIS timestep, matching the per-step vmap in the reference) -> leaky.
// One block per timestep l. dst layout: (L,B,Cout,HW).
// ---------------------------------------------------------------------------
__global__ void gen_stage(const float* __restrict__ src, int srcIsX, int Cin, int Cout,
                          const float* __restrict__ wgt, const float* __restrict__ bias,
                          const float* __restrict__ gma, const float* __restrict__ bta,
                          float* __restrict__ dst)
{
    int l = blockIdx.x;
    int t = threadIdx.x;
    int total = B_T * Cout * HW_T;
    for (int idx = t; idx < total; idx += 256) {
        int b  = idx / (Cout * HW_T);
        int co = (idx / HW_T) % Cout;
        int hw = idx & 255;
        int h = hw >> 4, wx = hw & 15;
        float acc = bias[co];
        for (int ci = 0; ci < Cin; ++ci) {
            for (int ky = 0; ky < 3; ++ky) {
                int h2 = h + ky - 1;
                if (h2 < 0 || h2 > 15) continue;
                for (int kx = 0; kx < 3; ++kx) {
                    int w2 = wx + kx - 1;
                    if (w2 < 0 || w2 > 15) continue;
                    float sv;
                    if (srcIsX)  // x: (B,CIN,H,W,L)
                        sv = src[(size_t)((b * Cin + ci) * HW_T + h2 * 16 + w2) * L_T + l];
                    else         // stage buf: (L,B,C,HW)
                        sv = src[(size_t)((l * B_T + b) * Cin + ci) * HW_T + h2 * 16 + w2];
                    acc += wgt[((co * Cin + ci) * 3 + ky) * 3 + kx] * sv;
                }
            }
        }
        dst[(size_t)((l * B_T + b) * Cout + co) * HW_T + hw] = acc;
    }
    __threadfence_block();
    __syncthreads();

    __shared__ float red[256], red2[256];
    __shared__ float scs[64], shs[64];
    for (int co = 0; co < Cout; ++co) {
        float s = 0.f, s2 = 0.f;
        for (int j = t; j < B_T * HW_T; j += 256) {
            int b = j >> 8, hw = j & 255;
            float v = dst[(size_t)((l * B_T + b) * Cout + co) * HW_T + hw];
            s += v; s2 += v * v;
        }
        red[t] = s; red2[t] = s2;
        __syncthreads();
        for (int off = 128; off > 0; off >>= 1) {
            if (t < off) { red[t] += red[t + off]; red2[t] += red2[t + off]; }
            __syncthreads();
        }
        if (t == 0) {
            float m   = red[0] * (1.f / 512.f);
            float var = red2[0] * (1.f / 512.f) - m * m;
            float inv = rsqrtf(var + 1e-5f);
            float sc  = gma[co] * inv;
            scs[co] = sc;
            shs[co] = bta[co] - m * sc;
        }
        __syncthreads();
    }
    for (int idx = t; idx < total; idx += 256) {
        int b  = idx / (Cout * HW_T);
        int co = (idx / HW_T) % Cout;
        int hw = idx & 255;
        float* p = &dst[(size_t)((l * B_T + b) * Cout + co) * HW_T + hw];
        *p = leakyf(*p * scs[co] + shs[co]);
    }
}

// ---------------------------------------------------------------------------
// Positional table (L,D): sin on even l, cos on odd l.
// ---------------------------------------------------------------------------
__global__ void pos_table_kernel(float* __restrict__ tab)
{
    int i = blockIdx.x * 256 + threadIdx.x;
    if (i >= L_T * D_T) return;
    int l = i / D_T, d = i % D_T;
    float inv = powf(10000.f, (float)(2 * (d / 2)) / (float)D_T);
    float ang = (float)l / inv;
    tab[i] = (l & 1) ? cosf(ang) : sinf(ang);
}

__global__ void pos_add_kernel(float* __restrict__ z, const float* __restrict__ tab)
{
    int i = blockIdx.x * 256 + threadIdx.x;   // 1,048,576 threads, z:(L,B,D,HW)
    int l = i >> 15;
    int d = (i >> 8) & 63;
    z[i] += tab[l * D_T + d];
}

// ---------------------------------------------------------------------------
// LayerNorm over (D,H,W) per (l,b); writes f16 A-matrix for the next GEMM.
// mode 0: rows m=(b*HW+hw)*L+l, cols d  (KV projection A)
// mode 1: rows m=(b*D+d)*HW+hw,  cols l  (FF A)
// ---------------------------------------------------------------------------
__global__ void ln_kernel(const float* __restrict__ z, const float* __restrict__ g,
                          const float* __restrict__ bb, _Float16* __restrict__ outh, int mode)
{
    int lb = blockIdx.x;             // = l*2 + b
    int l = lb >> 1, b = lb & 1;
    int t = threadIdx.x;
    const float* base = z + (size_t)lb * 16384;
    float s = 0.f, s2 = 0.f;
    for (int j = t; j < 16384; j += 256) { float v = base[j]; s += v; s2 += v * v; }
    __shared__ float red[256], red2[256];
    __shared__ float sm, sinv;
    red[t] = s; red2[t] = s2;
    __syncthreads();
    for (int off = 128; off > 0; off >>= 1) {
        if (t < off) { red[t] += red[t + off]; red2[t] += red2[t + off]; }
        __syncthreads();
    }
    if (t == 0) {
        float m   = red[0] * (1.f / 16384.f);
        float var = red2[0] * (1.f / 16384.f) - m * m;
        sm = m; sinv = rsqrtf(var + 1e-5f);
    }
    __syncthreads();
    float m = sm, inv = sinv;
    for (int j = t; j < 16384; j += 256) {
        int d = j >> 8, hw = j & 255;
        float v = (base[j] - m) * inv * g[j] + bb[j];
        size_t o;
        if (mode == 0) o = (size_t)((b * HW_T + hw) * L_T + l) * D_T + d;
        else           o = (size_t)((b * D_T + d) * HW_T + hw) * L_T + l;
        outh[o] = (_Float16)v;
    }
}

// ---------------------------------------------------------------------------
// Fold (c1w K/V block) x (wik/wiv) into one effective 64x(4heads*{K16,V16})=64x128
// matrix (f16, stored transposed [n][c]) plus effective f32 bias[128].
// Q block and c2 bias are mathematically dead (cancel inside softmax).
// ---------------------------------------------------------------------------
__global__ void wkv_prep(const float* __restrict__ c1w, const float* __restrict__ c1b,
                         const float* __restrict__ wik, const float* __restrict__ wiv,
                         int layer, _Float16* __restrict__ Wt, float* __restrict__ bkv)
{
    int t = threadIdx.x;
    for (int e = t; e < 128 * 64; e += 256) {
        int n = e >> 6, cp = e & 63;
        int hh = n >> 5, sel = (n >> 4) & 1, d = n & 15;
        const float* wsel = sel ? wiv : wik;
        int obase = 64 + 64 * sel;   // K rows 64..127, V rows 128..191 of c1w
        float s = 0.f;
        for (int c2 = 0; c2 < 64; ++c2) {
            float cw = c1w[(size_t)(((layer * HD_T + hh) * 192) + obase + c2) * 64 + cp];
            float wv = wsel[(size_t)((layer * HD_T + hh) * 64 + c2) * 16 + d];
            s += cw * wv;
        }
        Wt[n * 64 + cp] = (_Float16)s;
    }
    if (t < 128) {
        int n = t;
        int hh = n >> 5, sel = (n >> 4) & 1, d = n & 15;
        const float* wsel = sel ? wiv : wik;
        int obase = 64 + 64 * sel;
        float s = 0.f;
        for (int c2 = 0; c2 < 64; ++c2)
            s += c1b[(layer * HD_T + hh) * 192 + obase + c2] *
                 wsel[(size_t)((layer * HD_T + hh) * 64 + c2) * 16 + d];
        bkv[n] = s;
    }
}

// ---------------------------------------------------------------------------
// Generic WMMA GEMM: C[M,N] = A[M,K](f16,row-major) * Bt[N,K](f16)^T + bias[N].
// One 16x16 C tile per wave; K in steps of 32 via v_wmma_f32_16x16x32_f16.
// Per-lane operand packing follows the CDNA5 ISA VGPR layout tables:
//   A 16x32: lane m=lane&15, halves h0..7 -> K = hi*8 + h, h8..15 -> K = 16+hi*8+(h-8)
//   B 32x16: lane n=lane&15, halves h0..7 -> K = hi*8 + h, h8..15 -> K = 16+hi*8+(h-8)  (hi = lane>>4)
// Both are contiguous 16B chunks -> plain uint4 loads.
// ---------------------------------------------------------------------------
__global__ void gemm_wmma_kernel(const _Float16* __restrict__ A, const _Float16* __restrict__ Bt,
                                 const float* __restrict__ bias, float* __restrict__ Cf,
                                 _Float16* __restrict__ Ch, int M, int N, int K, int doLeaky)
{
    int wave = (int)((blockIdx.x * blockDim.x + threadIdx.x) >> 5);
    int lane = threadIdx.x & 31;
    int tilesN = N >> 4;
    int nt = (M >> 4) * tilesN;
    if (wave >= nt) return;                 // whole-wave uniform exit: EXEC all-1s inside
    int tm = wave / tilesN, tn = wave - tm * tilesN;
    int hi = lane >> 4, ln = lane & 15;
    const _Float16* arow = A  + (size_t)(tm * 16 + ln) * K + hi * 8;
    const _Float16* brow = Bt + (size_t)(tn * 16 + ln) * K + hi * 8;
    v8f acc = {};
    for (int kk = 0; kk < K; kk += 32) {
        H16 a, bm;
        a.q[0]  = *(const uint4*)(arow + kk);
        a.q[1]  = *(const uint4*)(arow + kk + 16);
        bm.q[0] = *(const uint4*)(brow + kk);
        bm.q[1] = *(const uint4*)(brow + kk + 16);
        acc = __builtin_amdgcn_wmma_f32_16x16x32_f16(false, a.v, false, bm.v,
                                                     (short)0, acc, false, false);
    }
    int n = tn * 16 + ln;
    float bv = bias ? bias[n] : 0.f;
#pragma unroll
    for (int r = 0; r < 8; ++r) {
        int m = tm * 16 + r + hi * 8;       // C/D layout: VGPR r -> row r (+8 for hi lanes)
        float v = acc[r] + bv;
        if (doLeaky) v = leakyf(v);
        if (Ch) Ch[(size_t)m * N + n] = (_Float16)v;
        else    Cf[(size_t)m * N + n] = v;
    }
}

// ---------------------------------------------------------------------------
// Fused attention tail per (b,head,16-position tile):
//   SK = conv5x5_nobias(K) over (h,w) per l; softmax over l; O = sum_l w_l * V_l.
// (Query/bias terms cancel in the softmax -> output is query-independent.)
// CKV layout: [m=(b*HW+hw)*L+l][n=hh*32 + {0..15 K | 16..31 V}]
// ---------------------------------------------------------------------------
__global__ void attn_conv_softmax(const float* __restrict__ ckv,
                                  const float* __restrict__ c2wL,  // already + layer offset
                                  float* __restrict__ oheads)
{
    __shared__ float wsh[DK_T * DK_T * 25];   // 6400 floats: [d][c][ky][kx]
    __shared__ float skb[256 * 33];
    int bid = blockIdx.x;                     // 0..127
    int b  = bid >> 6;
    int hh = (bid >> 4) & 3;
    int pg = bid & 15;
    int t  = threadIdx.x;
    for (int i = t; i < 6400; i += 256) wsh[i] = c2wL[hh * 6400 + i];
    __syncthreads();

    int d  = t & 15;
    int hw = pg * 16 + (t >> 4);
    int h = hw >> 4, wx = hw & 15;
    float* sk = &skb[t * 33];

    for (int l = 0; l < L_T; ++l) {
        float acc = 0.f;
        for (int ky = 0; ky < 5; ++ky) {
            int h2 = h + ky - 2;
            if (h2 < 0 || h2 > 15) continue;
            for (int kx = 0; kx < 5; ++kx) {
                int w2 = wx + kx - 2;
                if (w2 < 0 || w2 > 15) continue;
                const float* kp = ckv + (size_t)((b * HW_T + h2 * 16 + w2) * L_T + l) * 128 + hh * 32;
                const float* wp = &wsh[d * 400 + ky * 5 + kx];
                float a2 = 0.f;
                for (int c = 0; c < 16; ++c) a2 += wp[c * 25] * kp[c];
                acc += a2;
            }
        }
        sk[l] = acc;
    }
    float mx = -3.4e38f;
    for (int l = 0; l < L_T; ++l) mx = fmaxf(mx, sk[l]);
    float ssum = 0.f;
    for (int l = 0; l < L_T; ++l) { float e = expf(sk[l] - mx); sk[l] = e; ssum += e; }
    float rs = 1.f / ssum;
    float o = 0.f;
    const float* vp = ckv + (size_t)((b * HW_T + hw) * L_T) * 128 + hh * 32 + 16 + d;
    for (int l = 0; l < L_T; ++l) o += sk[l] * rs * vp[(size_t)l * 128];
    oheads[(b * D_T + hh * 16 + d) * HW_T + hw] = o;
}

// att[b,o,hw] = sum_c oheads[b,c,hw] * w0[c,o]
__global__ void w0_kernel(const float* __restrict__ oh, const float* __restrict__ w0L,
                          float* __restrict__ att)
{
    int i = blockIdx.x * 256 + threadIdx.x;   // 32768
    int b = i >> 14, o = (i >> 8) & 63, hw = i & 255;
    float s = 0.f;
    for (int c = 0; c < 64; ++c)
        s += oh[(b * D_T + c) * HW_T + hw] * w0L[c * 64 + o];
    att[i] = s;
}

// z[l,b,d,hw] += att[b,d,hw]  (attention output is query-step independent)
__global__ void res_bcast(float* __restrict__ z, const float* __restrict__ att)
{
    int i = blockIdx.x * 256 + threadIdx.x;
    z[i] += att[i & 32767];
}

// z[l,b,d,hw] += C2[(b*64+d)*256+hw][l]
__global__ void res_ff(float* __restrict__ z, const float* __restrict__ c2)
{
    int i = blockIdx.x * 256 + threadIdx.x;
    int l = i >> 15, rem = i & 32767;
    z[i] += c2[(size_t)rem * L_T + l];
}

__global__ void ffw_prep(const float* __restrict__ w1, const float* __restrict__ w2,
                         _Float16* __restrict__ h1, _Float16* __restrict__ h2)
{
    int i = blockIdx.x * 256 + threadIdx.x;
    if (i < 4096)      h1[i] = (_Float16)w1[i];           // (FF,L)  = Bt[n=f][k=l]
    else if (i < 8192) h2[i - 4096] = (_Float16)w2[i - 4096]; // (L,FF) = Bt[n=l][k=f]
}

// out[(hw*L+l), b, d] = z[l,b,d,hw]
__global__ void out_kernel(const float* __restrict__ z, float* __restrict__ out)
{
    int i = blockIdx.x * 256 + threadIdx.x;
    int l = i >> 15, rem = i & 32767;
    int b = rem >> 14, d = (rem >> 8) & 63, hw = rem & 255;
    out[(size_t)(hw * L_T + l) * (B_T * D_T) + b * D_T + d] = z[i];
}

// ---------------------------------------------------------------------------
extern "C" void kernel_launch(void* const* d_in, const int* in_sizes, int n_in,
                              void* d_out, int out_size, void* d_ws, size_t ws_size,
                              hipStream_t stream)
{
    const float* x     = (const float*)d_in[0];
    const float* gw[4] = { (const float*)d_in[1],  (const float*)d_in[5],
                           (const float*)d_in[9],  (const float*)d_in[13] };
    const float* gb[4] = { (const float*)d_in[2],  (const float*)d_in[6],
                           (const float*)d_in[10], (const float*)d_in[14] };
    const float* gg[4] = { (const float*)d_in[3],  (const float*)d_in[7],
                           (const float*)d_in[11], (const float*)d_in[15] };
    const float* ge[4] = { (const float*)d_in[4],  (const float*)d_in[8],
                           (const float*)d_in[12], (const float*)d_in[16] };
    const float* ln1g = (const float*)d_in[17];
    const float* ln1b = (const float*)d_in[18];
    const float* c1w  = (const float*)d_in[19];
    const float* c1b  = (const float*)d_in[20];
    const float* c2w  = (const float*)d_in[21];
    // d_in[22] (c2_b) and d_in[23] (wiq) cancel inside the softmax -> unused.
    const float* wik  = (const float*)d_in[24];
    const float* wiv  = (const float*)d_in[25];
    const float* w0   = (const float*)d_in[26];
    const float* ln2g = (const float*)d_in[27];
    const float* ln2b = (const float*)d_in[28];
    const float* ffw1 = (const float*)d_in[29];
    const float* ffb1 = (const float*)d_in[30];
    const float* ffw2 = (const float*)d_in[31];
    const float* ffb2 = (const float*)d_in[32];

    // Workspace carve-up (floats); f16 regions 16B aligned. ~25 MB total.
    float* ws   = (float*)d_ws;
    float* Z    = ws;                       // (L,B,64,HW)          1048576 f
    float* G1   = Z   + 1048576;            // gen stage1           65536 f
    float* G2   = G1  + 65536;              // gen stage2           98304 f
    float* G3   = G2  + 98304;              // gen stage3           131072 f
    float* TAB  = G3  + 131072;             // pos table            2048 f
    _Float16* ZNH  = (_Float16*)(TAB + 2048);   // 1048576 h (aliased for ln1/ln2 A)
    float*    CKV  = (float*)(ZNH + 1048576);   // 2097152 f (aliased with FF2 out C2)
    _Float16* HDNH = (_Float16*)(CKV + 2097152);// 4194304 h
    float*    OH   = (float*)(HDNH + 4194304);  // 32768 f
    float*    ATT  = OH + 32768;                // 32768 f
    _Float16* WKVT = (_Float16*)(ATT + 32768);  // 8192 h
    float*    BKV  = (float*)(WKVT + 8192);     // 128 f
    _Float16* FW1H = (_Float16*)(BKV + 128);    // 4096 h
    _Float16* FW2H = FW1H + 4096;               // 4096 h
    float*    C2   = CKV;                       // lifetime-disjoint alias
    _Float16* A2H  = ZNH;                       // lifetime-disjoint alias

    // Feature generator (per-timestep conv+BN+leaky x4), stage4 writes Z directly.
    gen_stage<<<32, 256, 0, stream>>>(x,  1, 6, 4,  gw[0], gb[0], gg[0], ge[0], G1);
    gen_stage<<<32, 256, 0, stream>>>(G1, 0, 4, 6,  gw[1], gb[1], gg[1], ge[1], G2);
    gen_stage<<<32, 256, 0, stream>>>(G2, 0, 6, 8,  gw[2], gb[2], gg[2], ge[2], G3);
    gen_stage<<<32, 256, 0, stream>>>(G3, 0, 8, 64, gw[3], gb[3], gg[3], ge[3], Z);
    pos_table_kernel<<<8, 256, 0, stream>>>(TAB);
    pos_add_kernel<<<4096, 256, 0, stream>>>(Z, TAB);

    for (int li = 0; li < 2; ++li) {
        // --- attention branch ---
        ln_kernel<<<64, 256, 0, stream>>>(Z, ln1g + li * 16384, ln1b + li * 16384, ZNH, 0);
        wkv_prep<<<1, 256, 0, stream>>>(c1w, c1b, wik, wiv, li, WKVT, BKV);
        // K,V for all heads: [16384 x 64] x [64 x 128] -> CKV (WMMA)
        gemm_wmma_kernel<<<1024, 256, 0, stream>>>(ZNH, WKVT, BKV, CKV, nullptr,
                                                   16384, 128, 64, 0);
        attn_conv_softmax<<<128, 256, 0, stream>>>(CKV, c2w + li * HD_T * 6400, OH);
        w0_kernel<<<128, 256, 0, stream>>>(OH, w0 + li * 4096, ATT);
        res_bcast<<<4096, 256, 0, stream>>>(Z, ATT);

        // --- feed-forward over time axis ---
        ln_kernel<<<64, 256, 0, stream>>>(Z, ln2g + li * 16384, ln2b + li * 16384, A2H, 1);
        ffw_prep<<<32, 256, 0, stream>>>(ffw1 + li * 4096, ffw2 + li * 4096, FW1H, FW2H);
        // FF1: [32768 x 32] x [32 x 128] + b, leaky, f16 out (WMMA)
        gemm_wmma_kernel<<<2048, 256, 0, stream>>>(A2H, FW1H, ffb1 + li * 128,
                                                   nullptr, HDNH, 32768, 128, 32, 1);
        // FF2: [32768 x 128] x [128 x 32] + b, f32 out (WMMA)
        gemm_wmma_kernel<<<512, 256, 0, stream>>>(HDNH, FW2H, ffb2 + li * 32,
                                                  C2, nullptr, 32768, 32, 128, 0);
        res_ff<<<4096, 256, 0, stream>>>(Z, C2);
    }

    out_kernel<<<4096, 256, 0, stream>>>(Z, (float*)d_out);

    (void)in_sizes; (void)n_in; (void)out_size; (void)ws_size;
}
